// FeatureEncoder_72928544686205
// MI455X (gfx1250) — compile-verified, matrix operand order checked by reading, not run on hardware
//
#include <hip/hip_runtime.h>

// ---------------- problem constants ----------------
constexpr int kN  = 4096;
constexpr int kE  = 65536;
constexpr int kEN = kE + kN;     // edges + self loops
constexpr int kB  = 32;
constexpr int kD  = 78;
constexpr int kH  = 10;
constexpr int kD2 = kD * kH;     // 780
constexpr int kFF = 2048;
constexpr int kL  = 16885;
// conv pipeline lengths
constexpr int kL1 = kL - 7;        // 16878
constexpr int kP1 = kL1 / 3;       // 5626
constexpr int kL2 = kP1 - 7;       // 5619
constexpr int kP2 = kL2 / 3;       // 1873
constexpr int kL3 = kP2 - 7;       // 1866
constexpr int kP3 = kL3 / 3;       // 622  (128*622 = 79616)

typedef __bf16 bf16;
typedef __attribute__((ext_vector_type(16))) __bf16 v16bf;
typedef __attribute__((ext_vector_type(8)))  float  v8f;

// order-preserving float <-> uint encoding for atomicMax on floats
__device__ __forceinline__ unsigned fenc(float f) {
    unsigned u = __float_as_uint(f);
    return (u & 0x80000000u) ? ~u : (u | 0x80000000u);
}
__device__ __forceinline__ float fdec(unsigned k) {
    unsigned u = (k & 0x80000000u) ? (k & 0x7FFFFFFFu) : ~k;
    return __uint_as_float(u);
}
#define ENC_NEG_INF 0x007FFFFFu   // fenc(-inf)

// ================= WMMA bf16 GEMM =================
// C[M,Nc] = act( alpha * A[M,K] * op(B) + bias ), op(B)=B^T if transB (B is [Nc,K]),
// else B is [K,Nc]. A,B fp32 in global, staged to LDS as bf16.
// Block tile 128x128, 8 waves, wave tile 32x64 = 2x4 WMMA fragments (8 wmma/K-step).
// REQUIREMENT: K and all leading dims are even (true for every call site here),
// so pair-clamped float2 loads are always in-bounds and 8B-aligned -> branchless staging.
#define BM 128
#define BN 128
#define BK 32

__global__ __launch_bounds__(256) void gemm_wmma_bf16(
    const float* __restrict__ A, int lda,
    const float* __restrict__ B, int ldb,
    const float* __restrict__ bias,
    float* __restrict__ C, int ldc,
    int M, int Nc, int K, float alpha, int transB, int act)
{
    __shared__ bf16 sA [BM * BK];   // [m][k]
    __shared__ bf16 sBT[BN * BK];   // [n][k]  (B transposed: fragment reads contiguous in k)

    const int tid  = threadIdx.x;
    const int lane = tid & 31;
    const int wave = tid >> 5;       // 0..7
    const int wm   = wave >> 1;      // 0..3 -> 32-row slab
    const int wn   = wave & 1;       // 0..1 -> 64-col slab
    const int h    = lane >> 4;      // lane half
    const int r    = lane & 15;
    const int blockM = blockIdx.y * BM;
    const int blockN = blockIdx.x * BN;

    v8f acc[2][4];
    #pragma unroll
    for (int i = 0; i < 2; ++i)
        #pragma unroll
        for (int j = 0; j < 4; ++j)
            #pragma unroll
            for (int v = 0; v < 8; ++v) acc[i][j][v] = 0.0f;

    for (int kt = 0; kt < K; kt += BK) {
        // ---------- stage A tile (128x32), branchless two-phase ----------
        {
            float2 ra[8];
            bool   pa[8];
            #pragma unroll
            for (int t = 0; t < 8; ++t) {
                int idx = tid + t * 256;            // 0..2047
                int row = idx >> 4, grp = idx & 15; // 16 float2 per row
                int gm = blockM + row;
                int gk = kt + grp * 2;
                int cm = (gm < M) ? gm : (M - 1);
                int ck = (gk < K) ? gk : (K - 2);   // K even -> pair stays in range
                ra[t] = *(const float2*)(A + (long)cm * lda + ck);
                pa[t] = (gm < M) & (gk < K);
            }
            #pragma unroll
            for (int t = 0; t < 8; ++t) {
                int idx = tid + t * 256;
                int row = idx >> 4, grp = idx & 15;
                float x = pa[t] ? ra[t].x : 0.0f;
                float y = pa[t] ? ra[t].y : 0.0f;
                bf16* d = &sA[row * BK + grp * 2];
                d[0] = (bf16)x; d[1] = (bf16)y;
            }
        }

        // ---------- stage B tile into sBT[n][k] ----------
        if (transB) {
            float2 rb[8];
            bool   pb[8];
            #pragma unroll
            for (int t = 0; t < 8; ++t) {
                int idx = tid + t * 256;
                int n = idx >> 4, grp = idx & 15;
                int gn = blockN + n;
                int gk = kt + grp * 2;
                int cn = (gn < Nc) ? gn : (Nc - 1);
                int ck = (gk < K) ? gk : (K - 2);
                rb[t] = *(const float2*)(B + (long)cn * ldb + ck);
                pb[t] = (gn < Nc) & (gk < K);
            }
            #pragma unroll
            for (int t = 0; t < 8; ++t) {
                int idx = tid + t * 256;
                int n = idx >> 4, grp = idx & 15;
                float x = pb[t] ? rb[t].x : 0.0f;
                float y = pb[t] ? rb[t].y : 0.0f;
                bf16* d = &sBT[n * BK + grp * 2];
                d[0] = (bf16)x; d[1] = (bf16)y;
            }
        } else {
            // B is [K,Nc]; read coalesced along n (scalar, clamped), scatter into sBT
            float rb[16];
            bool  pb[16];
            #pragma unroll
            for (int t = 0; t < 16; ++t) {
                int idx = tid + t * 256;
                int k = idx >> 7, n = idx & 127;
                int gk = kt + k, gn = blockN + n;
                int ck = (gk < K) ? gk : (K - 1);
                int cn = (gn < Nc) ? gn : (Nc - 1);
                rb[t] = B[(long)ck * ldb + cn];
                pb[t] = (gk < K) & (gn < Nc);
            }
            #pragma unroll
            for (int t = 0; t < 16; ++t) {
                int idx = tid + t * 256;
                int k = idx >> 7, n = idx & 127;
                sBT[n * BK + k] = (bf16)(pb[t] ? rb[t] : 0.0f);
            }
        }

        // hint next K tile into cache (-> global_prefetch_b8)
        if (kt + BK < K) {
            __builtin_prefetch(A + (long)(blockM + (tid & 127)) * lda + kt + BK, 0, 0);
        }
        __syncthreads();

        // ---- fragments (ISA 7.12.2 wave32 layouts) ----
        // element i of a 16-elem bf16 fragment: k = 8*h + (i<8 ? i : i+8)
        // -> two contiguous 8-element (16B) LDS reads per fragment (ds_load_b128)
        v16bf aF[2], bF[4];
        #pragma unroll
        for (int mi = 0; mi < 2; ++mi) {
            const bf16* base = &sA[(wm * 32 + mi * 16 + r) * BK + 8 * h];
            #pragma unroll
            for (int i = 0; i < 8; ++i) aF[mi][i] = base[i];
            #pragma unroll
            for (int i = 0; i < 8; ++i) aF[mi][8 + i] = base[16 + i];
        }
        #pragma unroll
        for (int ni = 0; ni < 4; ++ni) {
            const bf16* base = &sBT[(wn * 64 + ni * 16 + r) * BK + 8 * h];
            #pragma unroll
            for (int i = 0; i < 8; ++i) bF[ni][i] = base[i];
            #pragma unroll
            for (int i = 0; i < 8; ++i) bF[ni][8 + i] = base[16 + i];
        }
        #pragma unroll
        for (int mi = 0; mi < 2; ++mi)
            #pragma unroll
            for (int ni = 0; ni < 4; ++ni)
                acc[mi][ni] = __builtin_amdgcn_wmma_f32_16x16x32_bf16(
                    false, aF[mi], false, bF[ni], (short)0, acc[mi][ni], false, false);
        __syncthreads();
    }

    // ---- epilogue: D layout -> m = v + 8*h, n = r ----
    #pragma unroll
    for (int mi = 0; mi < 2; ++mi) {
        #pragma unroll
        for (int ni = 0; ni < 4; ++ni) {
            int gn = blockN + wn * 64 + ni * 16 + r;
            if (gn >= Nc) continue;
            float bb = bias ? bias[gn] : 0.0f;
            #pragma unroll
            for (int v = 0; v < 8; ++v) {
                int gm = blockM + wm * 32 + mi * 16 + 8 * h + v;
                if (gm >= M) continue;
                float f = acc[mi][ni][v] * alpha + bb;
                if (act) f = fmaxf(f, 0.0f);
                C[(long)gm * ldc + gn] = f;
            }
        }
    }
}

// ================= elementwise / reduction kernels =================
__global__ void fill_f32(float* p, float v, long n) {
    long i = (long)blockIdx.x * 256 + threadIdx.x;
    if (i < n) p[i] = v;
}
__global__ void fill_u32(unsigned* p, unsigned v, long n) {
    long i = (long)blockIdx.x * 256 + threadIdx.x;
    if (i < n) p[i] = v;
}

__global__ __launch_bounds__(256) void softmax_rows(float* x, int cols) {
    __shared__ float red[256];
    float* p = x + (long)blockIdx.x * cols;
    int t = threadIdx.x;
    float mx = -3.4e38f;
    for (int j = t; j < cols; j += 256) mx = fmaxf(mx, p[j]);
    red[t] = mx; __syncthreads();
    for (int s = 128; s > 0; s >>= 1) { if (t < s) red[t] = fmaxf(red[t], red[t + s]); __syncthreads(); }
    mx = red[0]; __syncthreads();
    float sum = 0.0f;
    for (int j = t; j < cols; j += 256) { float e = __expf(p[j] - mx); p[j] = e; sum += e; }
    red[t] = sum; __syncthreads();
    for (int s = 128; s > 0; s >>= 1) { if (t < s) red[t] += red[t + s]; __syncthreads(); }
    float inv = 1.0f / red[0];
    for (int j = t; j < cols; j += 256) p[j] *= inv;
}

// out = LayerNorm(xin + res) * w + b   (res may be nullptr)
__global__ __launch_bounds__(256) void ln_residual(
    const float* __restrict__ xin, const float* __restrict__ res,
    const float* __restrict__ w, const float* __restrict__ b,
    float* __restrict__ out, int d)
{
    __shared__ float red[256];
    int t = threadIdx.x;
    const float* px = xin + (long)blockIdx.x * d;
    const float* pr = res ? res + (long)blockIdx.x * d : nullptr;
    float* po = out + (long)blockIdx.x * d;
    float s = 0.0f, s2 = 0.0f;
    for (int j = t; j < d; j += 256) {
        float y = px[j] + (pr ? pr[j] : 0.0f);
        s += y; s2 += y * y;
    }
    red[t] = s; __syncthreads();
    for (int k = 128; k > 0; k >>= 1) { if (t < k) red[t] += red[t + k]; __syncthreads(); }
    s = red[0]; __syncthreads();
    red[t] = s2; __syncthreads();
    for (int k = 128; k > 0; k >>= 1) { if (t < k) red[t] += red[t + k]; __syncthreads(); }
    s2 = red[0];
    float m = s / d;
    float var = s2 / d - m * m;
    float inv = rsqrtf(var + 1e-5f);
    for (int j = t; j < d; j += 256) {
        float y = px[j] + (pr ? pr[j] : 0.0f);
        po[j] = (y - m) * inv * w[j] + b[j];
    }
}

__global__ void bias_relu(const float* __restrict__ in, const float* __restrict__ bias,
                          float* __restrict__ out, long n, int d) {
    long i = (long)blockIdx.x * 256 + threadIdx.x;
    if (i < n) {
        float v = in[i] + bias[i % d];
        out[i] = fmaxf(v, 0.0f);
    }
}

// ================= GAT =================
__global__ void gat_scores(const float* __restrict__ xh, const float* __restrict__ aw_src,
                           const float* __restrict__ aw_dst, float* __restrict__ asrc,
                           float* __restrict__ adst) {
    int id = blockIdx.x * 256 + threadIdx.x;
    if (id >= kN * kH) return;
    int n = id / kH, h = id % kH;
    const float* p = xh + (long)n * kD2 + h * kD;
    float s1 = 0.0f, s2 = 0.0f;
    for (int d = 0; d < kD; ++d) { s1 += p[d] * aw_src[h * kD + d]; s2 += p[d] * aw_dst[h * kD + d]; }
    asrc[id] = s1; adst[id] = s2;
}

__global__ void gat_edge1(const int* __restrict__ ei, const float* __restrict__ asrc,
                          const float* __restrict__ adst, float* __restrict__ ebuf,
                          unsigned* __restrict__ mkey) {
    int e = blockIdx.x * 256 + threadIdx.x;
    if (e >= kEN) return;
    int s = (e < kE) ? ei[e] : (e - kE);
    int d = (e < kE) ? ei[kE + e] : (e - kE);
    for (int h = 0; h < kH; ++h) {
        float v = asrc[s * kH + h] + adst[d * kH + h];
        v = (v > 0.0f) ? v : 0.2f * v;                 // leaky_relu 0.2
        ebuf[(long)e * kH + h] = v;
        atomicMax(&mkey[d * kH + h], fenc(v));
    }
}

__global__ void gat_edge2(const int* __restrict__ ei, float* __restrict__ ebuf,
                          const unsigned* __restrict__ mkey, float* __restrict__ ssum) {
    int e = blockIdx.x * 256 + threadIdx.x;
    if (e >= kEN) return;
    int d = (e < kE) ? ei[kE + e] : (e - kE);
    for (int h = 0; h < kH; ++h) {
        float v = __expf(ebuf[(long)e * kH + h] - fdec(mkey[d * kH + h]));
        ebuf[(long)e * kH + h] = v;
        atomicAdd(&ssum[d * kH + h], v);
    }
}

__global__ void gat_edge3(const int* __restrict__ ei, const float* __restrict__ ebuf,
                          const float* __restrict__ ssum, const float* __restrict__ xh,
                          float* __restrict__ out) {
    long id = (long)blockIdx.x * 256 + threadIdx.x;
    if (id >= (long)kEN * kH) return;
    int h = (int)(id % kH);
    int e = (int)(id / kH);
    int s = (e < kE) ? ei[e] : (e - kE);
    int d = (e < kE) ? ei[kE + e] : (e - kE);
    float alpha = ebuf[id] / ssum[d * kH + h];
    const float* px = xh + (long)s * kD2 + h * kD;
    float* po = out + (long)d * kD2 + h * kD;
    for (int t = 0; t < kD; ++t) atomicAdd(&po[t], alpha * px[t]);
}

// ================= GCN =================
__global__ void deg_kernel(const int* __restrict__ ei, float* __restrict__ deg) {
    int e = blockIdx.x * 256 + threadIdx.x;
    if (e >= kEN) return;
    int d = (e < kE) ? ei[kE + e] : (e - kE);
    atomicAdd(&deg[d], 1.0f);
}

__global__ void gcn_scatter(const int* __restrict__ ei, const float* __restrict__ deg,
                            const float* __restrict__ xw, float* __restrict__ out) {
    long id = (long)blockIdx.x * 256 + threadIdx.x;
    if (id >= (long)kEN * kD2) return;
    int j = (int)(id % kD2);
    int e = (int)(id / kD2);
    int s = (e < kE) ? ei[e] : (e - kE);
    int d = (e < kE) ? ei[kE + e] : (e - kE);
    float coef = rsqrtf(deg[s]) * rsqrtf(deg[d]);
    atomicAdd(&out[(long)d * kD2 + j], coef * xw[(long)s * kD2 + j]);
}

// ================= pooling =================
__global__ void cnt_kernel(const int* __restrict__ batch, float* __restrict__ cnt) {
    int n = blockIdx.x * 256 + threadIdx.x;
    if (n < kN) atomicAdd(&cnt[batch[n]], 1.0f);
}
__global__ void pool_kernel(const float* __restrict__ x, const int* __restrict__ batch,
                            unsigned* __restrict__ gmaxk, float* __restrict__ gmean) {
    long id = (long)blockIdx.x * 256 + threadIdx.x;
    if (id >= (long)kN * kD2) return;
    int j = (int)(id % kD2);
    int n = (int)(id / kD2);
    int b = batch[n];
    float v = x[id];
    atomicMax(&gmaxk[b * kD2 + j], fenc(v));
    atomicAdd(&gmean[b * kD2 + j], v);
}
__global__ void build_g(const unsigned* __restrict__ gmaxk, const float* __restrict__ gmean,
                        const float* __restrict__ cnt, float* __restrict__ g) {
    int id = blockIdx.x * 256 + threadIdx.x;
    if (id >= kB * kD2) return;
    int j = id % kD2, b = id / kD2;
    g[b * (2 * kD2) + j]       = fdec(gmaxk[b * kD2 + j]);
    g[b * (2 * kD2) + kD2 + j] = gmean[b * kD2 + j] / fmaxf(cnt[b], 1.0f);
}

// ================= conv path =================
__global__ void conv1d_relu(const float* __restrict__ in, const float* __restrict__ w,
                            const float* __restrict__ bias, float* __restrict__ out,
                            int inC, int outC, int inLen, int outLen) {
    long id = (long)blockIdx.x * 256 + threadIdx.x;
    long total = (long)kB * outC * outLen;
    if (id >= total) return;
    int p = (int)(id % outLen);
    long t = id / outLen;
    int oc = (int)(t % outC);
    int b  = (int)(t / outC);
    float s = bias[oc];
    for (int ic = 0; ic < inC; ++ic) {
        const float* pi = in + ((long)b * inC + ic) * inLen + p;
        const float* pw = w + ((long)oc * inC + ic) * 8;
        #pragma unroll
        for (int k = 0; k < 8; ++k) s += pi[k] * pw[k];
    }
    out[id] = fmaxf(s, 0.0f);
}

__global__ void pool3_kernel(const float* __restrict__ in, float* __restrict__ out,
                             int C, int inLen, int outLen) {
    long id = (long)blockIdx.x * 256 + threadIdx.x;
    long total = (long)kB * C * outLen;
    if (id >= total) return;
    int p = (int)(id % outLen);
    long t = id / outLen;
    int c = (int)(t % C);
    int b = (int)(t / C);
    const float* pi = in + ((long)b * C + c) * inLen + 3 * p;
    out[id] = fmaxf(fmaxf(pi[0], pi[1]), pi[2]);
}

__global__ void concat_out(const float* __restrict__ g2, const float* __restrict__ t2,
                           float* __restrict__ out) {
    int id = blockIdx.x * 256 + threadIdx.x;
    if (id >= kB * 256) return;
    int j = id % 256, b = id / 256;
    out[id] = (j < 128) ? g2[b * 128 + j] : t2[b * 128 + (j - 128)];
}

// ================= host-side helpers =================
static inline unsigned gb(long n) { return (unsigned)((n + 255) / 256); }

static void gemm(hipStream_t st, const float* A, int lda, const float* B, int ldb,
                 const float* bias, float* C, int ldc, int M, int Nc, int K,
                 float alpha, int transB, int act) {
    dim3 grid((Nc + BN - 1) / BN, (M + BM - 1) / BM);
    gemm_wmma_bf16<<<grid, 256, 0, st>>>(A, lda, B, ldb, bias, C, ldc, M, Nc, K, alpha, transB, act);
}

// P: in_w,in_b,out_w,out_b,l1w,l1b,l2w,l2b,n1w,n1b,n2w,n2b
static void tlayer(hipStream_t st, const float* xin, int d, const float* const* P,
                   float* xout, float* qkv, float* att, float* t1, float* t2v, float* x1) {
    float scale = 1.0f / sqrtf((float)d);
    gemm(st, xin, d, P[0], d, P[1], qkv, 3 * d, kN, 3 * d, d, 1.0f, 1, 0);          // qkv
    gemm(st, qkv, 3 * d, qkv + d, 3 * d, nullptr, att, kN, kN, kN, d, scale, 1, 0); // q k^T / sqrt(d)
    softmax_rows<<<kN, 256, 0, st>>>(att, kN);
    gemm(st, att, kN, qkv + 2 * d, 3 * d, nullptr, t1, d, kN, d, kN, 1.0f, 0, 0);   // att v
    gemm(st, t1, d, P[2], d, P[3], t2v, d, kN, d, d, 1.0f, 1, 0);                   // out proj
    ln_residual<<<kN, 256, 0, st>>>(xin, t2v, P[8], P[9], x1, d);                   // ln1
    gemm(st, x1, d, P[4], d, P[5], qkv, kFF, kN, kFF, d, 1.0f, 1, 1);               // ff1 (relu)
    gemm(st, qkv, kFF, P[6], kFF, P[7], t2v, d, kN, d, kFF, 1.0f, 1, 0);            // ff2
    ln_residual<<<kN, 256, 0, st>>>(x1, t2v, P[10], P[11], xout, d);                // ln2
}

extern "C" void kernel_launch(void* const* d_in, const int* in_sizes, int n_in,
                              void* d_out, int out_size, void* d_ws, size_t ws_size,
                              hipStream_t stream) {
    (void)in_sizes; (void)n_in; (void)out_size; (void)ws_size;
    // ---- inputs (setup_inputs order) ----
    const float* x0    = (const float*)d_in[0];
    const int*   ei    = (const int*)d_in[1];
    const int*   batch = (const int*)d_in[2];
    const float* tg    = (const float*)d_in[3];
    const float* t1p[12]; for (int k = 0; k < 12; ++k) t1p[k] = (const float*)d_in[4 + k];
    const float* gat_w    = (const float*)d_in[16];
    const float* gat_asrc = (const float*)d_in[17];
    const float* gat_adst = (const float*)d_in[18];
    const float* gat_b    = (const float*)d_in[19];
    const float* t2p[12]; for (int k = 0; k < 12; ++k) t2p[k] = (const float*)d_in[20 + k];
    const float* gcn_w  = (const float*)d_in[32];
    const float* gcn_b  = (const float*)d_in[33];
    const float* fcg1_w = (const float*)d_in[34];
    const float* fcg1_b = (const float*)d_in[35];
    const float* fcg2_w = (const float*)d_in[36];
    const float* fcg2_b = (const float*)d_in[37];
    const float* c1_w = (const float*)d_in[38]; const float* c1_b = (const float*)d_in[39];
    const float* c2_w = (const float*)d_in[40]; const float* c2_b = (const float*)d_in[41];
    const float* c3_w = (const float*)d_in[42]; const float* c3_b = (const float*)d_in[43];
    const float* fct1_w = (const float*)d_in[44]; const float* fct1_b = (const float*)d_in[45];
    const float* fct2_w = (const float*)d_in[46]; const float* fct2_b = (const float*)d_in[47];

    // ---- workspace arena ----
    char* Wb = (char*)d_ws;
    size_t off = 0;
    auto alloc = [&](size_t elems) -> void* {
        void* p = Wb + off;
        off += (elems * 4 + 255) & ~(size_t)255;
        return p;
    };
    float* ATT  = (float*)alloc(17283072);             // max(4096*4096, 32*32*16878); reused as conv ping
    float* QKV  = (float*)alloc((size_t)kN * 2340);    // qkv / FF hidden / conv pong
    float* T1   = (float*)alloc((size_t)kN * kD2);     // att*v result, GAT/GCN accumulators
    float* T2V  = (float*)alloc((size_t)kN * kD2);
    float* X    = (float*)alloc((size_t)kN * kD2);
    float* X1   = (float*)alloc((size_t)kN * kD2);
    float* XH   = (float*)alloc((size_t)kN * kD2);
    float* ASRC = (float*)alloc((size_t)kN * kH);
    float* ADST = (float*)alloc((size_t)kN * kH);
    unsigned* MKEY = (unsigned*)alloc((size_t)kN * kH);
    float* SSUM = (float*)alloc((size_t)kN * kH);
    float* EBUF = (float*)alloc((size_t)kEN * kH);
    float* DEG  = (float*)alloc(kN);
    unsigned* GMAXK = (unsigned*)alloc((size_t)kB * kD2);
    float* GMEAN = (float*)alloc((size_t)kB * kD2);
    float* CNT  = (float*)alloc(kB);
    float* G    = (float*)alloc((size_t)kB * 2 * kD2);
    float* G1   = (float*)alloc((size_t)kB * 1500);
    float* G2   = (float*)alloc((size_t)kB * 128);
    float* TF1  = (float*)alloc((size_t)kB * 1024);
    float* TF2  = (float*)alloc((size_t)kB * 128);
    float* CA = ATT;   // conv ping (free after attention layers)
    float* CB = QKV;   // conv pong

    // ---- transformer 1 (d=78): x0 -> X ----
    tlayer(stream, x0, kD, t1p, X, QKV, ATT, T1, T2V, X1);

    // ---- GAT ----
    gemm(stream, X, kD, gat_w, kD, nullptr, XH, kD2, kN, kD2, kD, 1.0f, 1, 0);      // xh
    gat_scores<<<gb((long)kN * kH), 256, 0, stream>>>(XH, gat_asrc, gat_adst, ASRC, ADST);
    fill_u32<<<gb((long)kN * kH), 256, 0, stream>>>(MKEY, ENC_NEG_INF, (long)kN * kH);
    fill_f32<<<gb((long)kN * kH), 256, 0, stream>>>(SSUM, 0.0f, (long)kN * kH);
    gat_edge1<<<gb(kEN), 256, 0, stream>>>(ei, ASRC, ADST, EBUF, MKEY);
    gat_edge2<<<gb(kEN), 256, 0, stream>>>(ei, EBUF, MKEY, SSUM);
    fill_f32<<<gb((long)kN * kD2), 256, 0, stream>>>(T1, 0.0f, (long)kN * kD2);
    gat_edge3<<<gb((long)kEN * kH), 256, 0, stream>>>(ei, EBUF, SSUM, XH, T1);
    bias_relu<<<gb((long)kN * kD2), 256, 0, stream>>>(T1, gat_b, X, (long)kN * kD2, kD2);

    // ---- transformer 2 (d=780): X -> X ----
    tlayer(stream, X, kD2, t2p, X, QKV, ATT, T1, T2V, X1);

    // ---- GCN ----
    gemm(stream, X, kD2, gcn_w, kD2, nullptr, XH, kD2, kN, kD2, kD2, 1.0f, 1, 0);   // xw
    fill_f32<<<gb(kN), 256, 0, stream>>>(DEG, 0.0f, kN);
    deg_kernel<<<gb(kEN), 256, 0, stream>>>(ei, DEG);
    fill_f32<<<gb((long)kN * kD2), 256, 0, stream>>>(T1, 0.0f, (long)kN * kD2);
    gcn_scatter<<<gb((long)kEN * kD2), 256, 0, stream>>>(ei, DEG, XH, T1);
    bias_relu<<<gb((long)kN * kD2), 256, 0, stream>>>(T1, gcn_b, X, (long)kN * kD2, kD2);

    // ---- graph pooling + FC ----
    fill_u32<<<gb((long)kB * kD2), 256, 0, stream>>>(GMAXK, ENC_NEG_INF, (long)kB * kD2);
    fill_f32<<<gb((long)kB * kD2), 256, 0, stream>>>(GMEAN, 0.0f, (long)kB * kD2);
    fill_f32<<<gb(kB), 256, 0, stream>>>(CNT, 0.0f, kB);
    cnt_kernel<<<gb(kN), 256, 0, stream>>>(batch, CNT);
    pool_kernel<<<gb((long)kN * kD2), 256, 0, stream>>>(X, batch, GMAXK, GMEAN);
    build_g<<<gb((long)kB * kD2), 256, 0, stream>>>(GMAXK, GMEAN, CNT, G);
    gemm(stream, G, 2 * kD2, fcg1_w, 2 * kD2, fcg1_b, G1, 1500, kB, 1500, 2 * kD2, 1.0f, 1, 1);
    gemm(stream, G1, 1500, fcg2_w, 1500, fcg2_b, G2, 128, kB, 128, 1500, 1.0f, 1, 0);

    // ---- conv path on target_ge ----
    conv1d_relu<<<gb((long)kB * 32 * kL1), 256, 0, stream>>>(tg, c1_w, c1_b, CA, 1, 32, kL, kL1);
    pool3_kernel<<<gb((long)kB * 32 * kP1), 256, 0, stream>>>(CA, CB, 32, kL1, kP1);
    conv1d_relu<<<gb((long)kB * 64 * kL2), 256, 0, stream>>>(CB, c2_w, c2_b, CA, 32, 64, kP1, kL2);
    pool3_kernel<<<gb((long)kB * 64 * kP2), 256, 0, stream>>>(CA, CB, 64, kL2, kP2);
    conv1d_relu<<<gb((long)kB * 128 * kL3), 256, 0, stream>>>(CB, c3_w, c3_b, CA, 64, 128, kP2, kL3);
    pool3_kernel<<<gb((long)kB * 128 * kP3), 256, 0, stream>>>(CA, CB, 128, kL3, kP3);
    gemm(stream, CB, 79616, fct1_w, 79616, fct1_b, TF1, 1024, kB, 1024, 79616, 1.0f, 1, 1);
    gemm(stream, TF1, 1024, fct2_w, 1024, fct2_b, TF2, 128, kB, 128, 1024, 1.0f, 1, 0);

    // ---- final concat ----
    concat_out<<<gb((long)kB * 256), 256, 0, stream>>>(G2, TF2, (float*)d_out);
}
